// MLP_18545668784663
// MI455X (gfx1250) — matrix operand order repaired and checked by reading, not compiled, not run on hardware
//
#include <hip/hip_runtime.h>
#include <hip/hip_bf16.h>

// ---------------------------------------------------------------------------
// MoE MLP (8 experts, top-2, D=1024, H=4096, T=8192) for gfx1250 (MI455X).
// Route pairs by expert -> bf16-swizzle weights (134MB, L2-resident) ->
// fused block per 32-token expert tile (M=32): up-GEMM (WMMA bf16) + branch-
// free GELU -> LDS H (swizzled as down-phase A fragments) -> down-GEMM ->
// gated per-pair contributions -> gather (no atomics => deterministic).
// LDS: X 64KB + H 256KB = 320KB (full CDNA5 WGP allocation), 512 thr/16 waves.
// Each wave N-blocks 4 tiles so one A-fragment pair feeds 8 WMMAs.
// ---------------------------------------------------------------------------

#define NUM_EXPERTS 8
#define D_MODEL     1024
#define D_HIDDEN    4096
#define TOP_K       2
#define T_TOKENS    8192
#define PAIRS       (T_TOKENS * TOP_K)          // 16384

#define S_UP   (D_MODEL  / 32)                  // 32 K-steps, up proj
#define NT_UP  (D_HIDDEN / 16)                  // 256 N-tiles, up proj
#define S_DN   (D_HIDDEN / 32)                  // 128 K-steps, down proj
#define NT_DN  (D_MODEL  / 16)                  // 64 N-tiles, down proj

#define XBANK  (S_UP * 32 * 16)                 // 16384 bf16 per M-half
#define HBANK  (S_DN * 32 * 16)                 // 65536 bf16 per M-half

typedef __bf16 bf16;
typedef __attribute__((ext_vector_type(16))) __bf16 v16bf;
typedef __attribute__((ext_vector_type(8)))  __bf16 v8bf;
typedef __attribute__((ext_vector_type(8)))  float  v8f;

__device__ __forceinline__ bf16 f2bf(float f) {
    unsigned u = __builtin_bit_cast(unsigned, f);
    u += 0x7FFFu + ((u >> 16) & 1u);
    unsigned short h = (unsigned short)(u >> 16);
    return __builtin_bit_cast(bf16, h);
}

// Branch-free erf (Abramowitz-Stegun 7.1.26, |err| <= 1.5e-7): rcp + exp + FMAs.
__device__ __forceinline__ float erf_fast(float x) {
    float ax = __builtin_fabsf(x);
    float t  = __builtin_amdgcn_rcpf(__builtin_fmaf(0.3275911f, ax, 1.0f));
    float q  = __builtin_fmaf(1.061405429f, t, -1.453152027f);
    q = __builtin_fmaf(q, t, 1.421413741f);
    q = __builtin_fmaf(q, t, -0.284496736f);
    q = __builtin_fmaf(q, t, 0.254829592f);
    q = q * t;
    float e = __expf(-ax * ax);
    float r = __builtin_fmaf(-q, e, 1.0f);
    return __builtin_copysignf(r, x);
}

__device__ __forceinline__ float gelu_exact(float v) {
    return 0.5f * v * (1.0f + erf_fast(v * 0.70710678118654752f));
}

#define WMMA_BF16(A, B, C) __builtin_amdgcn_wmma_f32_16x16x32_bf16( \
        false, (A), false, (B), (short)0, (C), false, false)

// A-fragment (16x32 bf16): lane = half*16 + m, element e -> k = ((e>>3)<<4)|(half<<3)|(e&7)
//   => 8 consecutive k map to 8 consecutive elements (vectorizable staging).
// B-fragment (32x16 bf16): lane = half*16 + n, element e -> K = ((e>>3)<<4)|(half<<3)|(e&7), as A (stored swizzled).

// ---------------------------------------------------------------------------
// Routing
// ---------------------------------------------------------------------------
__global__ void route_zero_kernel(int* counts, int* fill) {
    int i = threadIdx.x;
    if (i < NUM_EXPERTS) { counts[i] = 0; fill[i] = 0; }
}

__global__ void route_count_kernel(const int* __restrict__ idx, int* counts) {
    int i = blockIdx.x * 256 + threadIdx.x;
    if (i < PAIRS) atomicAdd(&counts[idx[i] & 7], 1);
}

__global__ void route_offsets_kernel(const int* __restrict__ counts, int* offsets) {
    if (threadIdx.x == 0) {
        int a = 0;
        for (int e = 0; e < NUM_EXPERTS; ++e) { offsets[e] = a; a += counts[e]; }
    }
}

__global__ void route_scatter_kernel(const int* __restrict__ idx,
                                     const float* __restrict__ p,
                                     const int* __restrict__ offsets,
                                     int* fill,
                                     int* __restrict__ pair_token,
                                     float* __restrict__ pair_gate,
                                     int* __restrict__ pair_slot) {
    int i = blockIdx.x * 256 + threadIdx.x;
    if (i < PAIRS) {
        int e = idx[i] & 7;
        int pos = offsets[e] + atomicAdd(&fill[e], 1);
        pair_token[pos] = i >> 1;
        pair_gate[pos]  = p[i];
        pair_slot[i]    = pos;
    }
}

// ---------------------------------------------------------------------------
// Weight conversion f32 -> bf16 into per-lane B-fragment swizzle.
// dest index = ((e*S + s)*NT + nt)*512 + lane*16 + elem
// ---------------------------------------------------------------------------
__global__ void cvt_up_kernel(const float* __restrict__ w, bf16* __restrict__ o) {
    size_t i = (size_t)blockIdx.x * 256 + threadIdx.x;
    if (i >= (size_t)NUM_EXPERTS * D_MODEL * D_HIDDEN) return;
    int  el   = (int)(i & 15);
    int  lane = (int)((i >> 4) & 31);
    size_t blk = i >> 9;
    int  nt   = (int)(blk & (NT_UP - 1));
    size_t b2 = blk >> 8;
    int  s    = (int)(b2 & (S_UP - 1));
    int  e    = (int)(b2 >> 5);
    int n = lane & 15, half = lane >> 4;
    int gk = s * 32 + ((el >> 3) << 4) + half * 8 + (el & 7);
    int gn = nt * 16 + n;
    o[i] = f2bf(w[((size_t)e * D_MODEL + gk) * D_HIDDEN + gn]);
}

__global__ void cvt_dn_kernel(const float* __restrict__ w, bf16* __restrict__ o) {
    size_t i = (size_t)blockIdx.x * 256 + threadIdx.x;
    if (i >= (size_t)NUM_EXPERTS * D_HIDDEN * D_MODEL) return;
    int  el   = (int)(i & 15);
    int  lane = (int)((i >> 4) & 31);
    size_t blk = i >> 9;
    int  nt   = (int)(blk & (NT_DN - 1));
    size_t b2 = blk >> 6;
    int  s    = (int)(b2 & (S_DN - 1));
    int  e    = (int)(b2 >> 7);
    int n = lane & 15, half = lane >> 4;
    int gk = s * 32 + ((el >> 3) << 4) + half * 8 + (el & 7);
    int gn = nt * 16 + n;
    o[i] = f2bf(w[((size_t)e * D_HIDDEN + gk) * D_MODEL + gn]);
}

// ---------------------------------------------------------------------------
// Fused expert kernel: one block = one expert x one 32-token tile.
// 512 threads / 16 waves. Per K-step each wave reuses one A0/A1 pair across
// 4 B fragments => 8 WMMAs per 4 ds_load_b128 + 8 global_load_b128.
// ---------------------------------------------------------------------------
__global__ __launch_bounds__(512) void moe_fused_kernel(
    const float* __restrict__ x,
    const bf16*  __restrict__ wup,          // [E][S_UP][NT_UP][512] swizzled
    const bf16*  __restrict__ wdn,          // [E][S_DN][NT_DN][512] swizzled
    const int*   __restrict__ pair_token,
    const float* __restrict__ pair_gate,
    const int*   __restrict__ counts,
    const int*   __restrict__ offsets,
    float*       __restrict__ contrib)      // [PAIRS][D_MODEL]
{
    const int e    = blockIdx.y;
    const int tile = blockIdx.x;
    const int cnt  = counts[e];
    if (tile * 32 >= cnt) return;           // uniform exit
    const int off  = offsets[e];

    __shared__ __align__(32) bf16 Xs[2 * XBANK];   //  64 KB
    __shared__ __align__(32) bf16 Hs[2 * HBANK];   // 256 KB  (total = 320 KB WGP)

    const int tid  = threadIdx.x;
    const int wv   = tid >> 5;              // 0..15
    const int lane = tid & 31;

    // ---- Stage X tile (32 tokens) into LDS in A-fragment swizzle.
    // One group = 8 consecutive k = 8 consecutive fragment elements = 16B.
    for (int g = tid; g < 2 * 16 * S_UP * 4; g += 512) {   // 4096 groups
        int oct  = g & 1;
        int half = (g >> 1) & 1;
        int s    = (g >> 2) & (S_UP - 1);
        int m16  = (g >> 7) & 15;
        int bank = g >> 11;
        int row  = tile * 32 + bank * 16 + m16;
        int srow = row < cnt ? row : 0;
        int tok  = pair_token[off + srow];
        int k0   = oct * 16 + half * 8;
        const float4* xp = (const float4*)(x + (size_t)tok * D_MODEL + s * 32 + k0);
        float4 lo = xp[0];
        float4 hi = xp[1];
        v8bf v;
        v[0] = f2bf(lo.x); v[1] = f2bf(lo.y); v[2] = f2bf(lo.z); v[3] = f2bf(lo.w);
        v[4] = f2bf(hi.x); v[5] = f2bf(hi.y); v[6] = f2bf(hi.z); v[7] = f2bf(hi.w);
        int lnn = half * 16 + m16;
        *(v8bf*)(Xs + bank * XBANK + ((s * 32 + lnn) << 4) + oct * 8) = v;
    }
    __syncthreads();

    const int n  = lane & 15;
    const int mb = (lane >> 4) * 8;

    // ---- Up projection: H[32 x 4096] = gelu(X @ Wup[e]) ----
    for (int nc = 0; nc < 4; ++nc) {               // 4 passes x (16 waves x 4 nt)
        const int ntb = nc * 64 + wv * 4;           // base N-tile
        v8f acc0[4] = {}, acc1[4] = {};
        const bf16* bp = wup + ((size_t)e * S_UP * NT_UP + ntb) * 512 + lane * 16;
        #pragma unroll 2
        for (int s = 0; s < S_UP; ++s) {
            v16bf a0 = *(const v16bf*)(Xs + ((s * 32 + lane) << 4));
            v16bf a1 = *(const v16bf*)(Xs + XBANK + ((s * 32 + lane) << 4));
            const bf16* bs = bp + (size_t)s * (NT_UP * 512);
            #pragma unroll
            for (int j = 0; j < 4; ++j) {
                v16bf b = *(const v16bf*)(bs + j * 512);
                acc0[j] = WMMA_BF16(a0, b, acc0[j]);
                acc1[j] = WMMA_BF16(a1, b, acc1[j]);
            }
        }
        // GELU + scatter into Hs (down-phase A-fragment layout), both M halves.
        #pragma unroll
        for (int j = 0; j < 4; ++j) {
            const int hcol = (ntb + j) * 16 + n;
            const int sd   = hcol >> 5;
            const int k    = hcol & 31;
            const int half = (k >> 3) & 1;
            const int el   = ((k >> 4) << 3) | (k & 7);
            #pragma unroll
            for (int r = 0; r < 8; ++r) {
                int m = mb + r;
                int base = ((sd * 32 + half * 16 + m) << 4) + el;
                Hs[base]         = f2bf(gelu_exact(acc0[j][r]));
                Hs[HBANK + base] = f2bf(gelu_exact(acc1[j][r]));
            }
        }
    }
    __syncthreads();

    // ---- Down projection: Y[32 x 1024] = H @ Wdown[e], scaled by gate ----
    float g0[8], g1[8];
    #pragma unroll
    for (int r = 0; r < 8; ++r) {
        int r0 = tile * 32 + mb + r;
        int r1 = r0 + 16;
        g0[r] = (r0 < cnt) ? pair_gate[off + r0] : 0.0f;
        g1[r] = (r1 < cnt) ? pair_gate[off + r1] : 0.0f;
    }

    {
        const int ntb = wv * 4;                     // single pass: 16 waves x 4 nt
        v8f acc0[4] = {}, acc1[4] = {};
        const bf16* bp = wdn + ((size_t)e * S_DN * NT_DN + ntb) * 512 + lane * 16;
        #pragma unroll 2
        for (int s = 0; s < S_DN; ++s) {
            v16bf a0 = *(const v16bf*)(Hs + ((s * 32 + lane) << 4));
            v16bf a1 = *(const v16bf*)(Hs + HBANK + ((s * 32 + lane) << 4));
            const bf16* bs = bp + (size_t)s * (NT_DN * 512);
            #pragma unroll
            for (int j = 0; j < 4; ++j) {
                v16bf b = *(const v16bf*)(bs + j * 512);
                acc0[j] = WMMA_BF16(a0, b, acc0[j]);
                acc1[j] = WMMA_BF16(a1, b, acc1[j]);
            }
        }
        #pragma unroll
        for (int j = 0; j < 4; ++j) {
            const int nt = ntb + j;
            #pragma unroll
            for (int r = 0; r < 8; ++r) {
                int r0 = tile * 32 + mb + r;
                int r1 = r0 + 16;
                if (r0 < cnt)
                    contrib[(size_t)(off + r0) * D_MODEL + nt * 16 + n] = g0[r] * acc0[j][r];
                if (r1 < cnt)
                    contrib[(size_t)(off + r1) * D_MODEL + nt * 16 + n] = g1[r] * acc1[j][r];
            }
        }
    }
}

// ---------------------------------------------------------------------------
// Final gather: y[t] = contrib[slot(t,0)] + contrib[slot(t,1)]
// ---------------------------------------------------------------------------
__global__ void gather_y_kernel(const float* __restrict__ contrib,
                                const int* __restrict__ pair_slot,
                                float* __restrict__ y) {
    size_t i = (size_t)blockIdx.x * 256 + threadIdx.x;
    if (i >= (size_t)T_TOKENS * D_MODEL) return;
    int t = (int)(i >> 10);
    int d = (int)(i & (D_MODEL - 1));
    int s0 = pair_slot[t * 2 + 0];
    int s1 = pair_slot[t * 2 + 1];
    y[i] = contrib[(size_t)s0 * D_MODEL + d] + contrib[(size_t)s1 * D_MODEL + d];
}

// ---------------------------------------------------------------------------
extern "C" void kernel_launch(void* const* d_in, const int* in_sizes, int n_in,
                              void* d_out, int out_size, void* d_ws, size_t ws_size,
                              hipStream_t stream) {
    const float* x        = (const float*)d_in[0];
    const float* expert_p = (const float*)d_in[1];
    const int*   expert_i = (const int*)  d_in[2];
    const float* w_up     = (const float*)d_in[3];
    const float* w_down   = (const float*)d_in[4];
    float* y = (float*)d_out;

    char* ws = (char*)d_ws;
    bf16* wup_sw = (bf16*)ws;  ws += (size_t)NUM_EXPERTS * D_MODEL * D_HIDDEN * 2;
    bf16* wdn_sw = (bf16*)ws;  ws += (size_t)NUM_EXPERTS * D_HIDDEN * D_MODEL * 2;
    float* contrib = (float*)ws; ws += (size_t)PAIRS * D_MODEL * 4;
    int*   pair_token = (int*)ws;   ws += PAIRS * 4;
    float* pair_gate  = (float*)ws; ws += PAIRS * 4;
    int*   pair_slot  = (int*)ws;   ws += PAIRS * 4;
    int*   counts  = (int*)ws; ws += 256;
    int*   offsets = (int*)ws; ws += 256;
    int*   fill    = (int*)ws; ws += 256;

    route_zero_kernel<<<1, 32, 0, stream>>>(counts, fill);
    route_count_kernel<<<(PAIRS + 255) / 256, 256, 0, stream>>>(expert_i, counts);
    route_offsets_kernel<<<1, 32, 0, stream>>>(counts, offsets);
    route_scatter_kernel<<<(PAIRS + 255) / 256, 256, 0, stream>>>(
        expert_i, expert_p, offsets, fill, pair_token, pair_gate, pair_slot);

    const size_t wu_elems = (size_t)NUM_EXPERTS * D_MODEL * D_HIDDEN;
    cvt_up_kernel<<<(unsigned)((wu_elems + 255) / 256), 256, 0, stream>>>(w_up, wup_sw);
    cvt_dn_kernel<<<(unsigned)((wu_elems + 255) / 256), 256, 0, stream>>>(w_down, wdn_sw);

    dim3 grid(PAIRS / 32, NUM_EXPERTS);
    moe_fused_kernel<<<grid, 512, 0, stream>>>(
        x, wup_sw, wdn_sw, pair_token, pair_gate, counts, offsets, contrib);

    const size_t y_elems = (size_t)T_TOKENS * D_MODEL;
    gather_y_kernel<<<(unsigned)((y_elems + 255) / 256), 256, 0, stream>>>(
        contrib, pair_slot, y);
}
